// Population_80504866996715
// MI455X (gfx1250) — matrix operand order repaired and checked by reading, not compiled, not run on hardware
//
#include <hip/hip_runtime.h>
#include <hip/hip_bf16.h>

// ---------------- problem constants ----------------
#define BZn   1024
#define ROIn  116
#define DIMn  116
#define FLATn (ROIn*DIMn)      // 13456
#define FLATP 13472            // FLAT padded to multiple of 32 (K padding)
#define FLATR 13504            // FLAT padded to multiple of 64 (Bt row / N-strip padding)
#define MRn   (BZn*ROIn)       // 118784 rows for per-ROI GEMMs (multiple of 16)
#define CPn   128              // 116 padded to multiple of 32
#define TOPKn 8
#define INV_TEMP (1.0f/0.07f)
#define EPSn  1e-5f

typedef __attribute__((ext_vector_type(16))) __bf16 bf16x16;
typedef __attribute__((ext_vector_type(8)))  __bf16 bf16x8;
typedef __attribute__((ext_vector_type(8)))  float  f32x8;

__device__ __forceinline__ __bf16 f2bf(float f) {
    union { float f; unsigned u; } a; a.f = f;
    unsigned r = a.u + 0x7FFFu + ((a.u >> 16) & 1u);   // round-to-nearest-even
    unsigned short s = (unsigned short)(r >> 16);
    union { unsigned short s; __bf16 b; } o; o.s = s;
    return o.b;
}

__device__ __forceinline__ float leakyf(float v) { return v >= 0.0f ? v : 0.2f * v; }

// ---------------- WMMA GEMM: C[M,N] = epilogue(A_bf16[M,Kp] @ Bt_bf16[N,Kp]^T) --------
// Each wave computes a 16x64 strip (4 N-tiles) so the A fragment is reused 4x and four
// independent accumulator chains hide the WMMA->WMMA RAW hazard.
// A row-major [M, lda]; Bt row-major [Nrows, ldb] with Nrows a multiple of 64, rows
// beyond N zero-filled. Kp multiple of 32, M multiple of 16.
// mode: 0=scale+bias, 1=leaky(scale+bias), 2=relu(scale+bias), 4=addIn+leaky(acc+bias)
__global__ __launch_bounds__(256)
void wmma_gemm(const __bf16* __restrict__ A, int lda,
               const __bf16* __restrict__ Bt, int ldb,
               float* __restrict__ C, int ldc,
               int M, int N, int Kp,
               const float* __restrict__ bias,
               const float* __restrict__ addIn, int addLd,
               float scale, int mode)
{
    const int wave = threadIdx.x >> 5;
    const int lane = threadIdx.x & 31;
    const int tilesN = (N + 15) >> 4;
    const int strips = (tilesN + 3) >> 2;
    const long stripId = (long)blockIdx.x * 8 + wave;
    const long totStrips = (long)(M >> 4) * strips;
    if (stripId >= totStrips) return;
    const int tm = (int)(stripId / strips);
    const int ts = (int)(stripId % strips);

    const int m16 = lane & 15;       // A row / B column / C column within tile
    const int hi  = lane >> 4;       // half-wave selector
    const __bf16* __restrict__ arow = A  + (long)(tm * 16 + m16) * lda + hi * 16;
    const __bf16* __restrict__ brow = Bt + (long)(ts * 64 + m16) * ldb + hi * 16;
    const long bs = (long)16 * ldb;  // stride between N-tiles inside the strip

    f32x8 acc0 = {}, acc1 = {}, acc2 = {}, acc3 = {};
#pragma unroll 2
    for (int k = 0; k < Kp; k += 32) {
        // prefetch next k-block into WGP-level cache (speculative; OOB drops silently)
        __builtin_prefetch((const void*)(arow + k + 64), 0, 3);
        __builtin_prefetch((const void*)(brow + k + 64), 0, 3);
        __builtin_prefetch((const void*)(brow + bs + k + 64), 0, 3);
        __builtin_prefetch((const void*)(brow + 2 * bs + k + 64), 0, 3);
        __builtin_prefetch((const void*)(brow + 3 * bs + k + 64), 0, 3);
        union { bf16x8 h[2]; bf16x16 v; } a;
        a.v = *(const bf16x16*)(arow + k);          // K = hi*16 .. hi*16+15
        bf16x16 b0 = *(const bf16x16*)(brow + k);   // K = hi*16 .. hi*16+15
        bf16x16 b1 = *(const bf16x16*)(brow + bs + k);
        bf16x16 b2 = *(const bf16x16*)(brow + 2 * bs + k);
        bf16x16 b3 = *(const bf16x16*)(brow + 3 * bs + k);
        acc0 = __builtin_amdgcn_wmma_f32_16x16x32_bf16(false, a.v, false, b0, (short)0, acc0, false, false);
        acc1 = __builtin_amdgcn_wmma_f32_16x16x32_bf16(false, a.v, false, b1, (short)0, acc1, false, false);
        acc2 = __builtin_amdgcn_wmma_f32_16x16x32_bf16(false, a.v, false, b2, (short)0, acc2, false, false);
        acc3 = __builtin_amdgcn_wmma_f32_16x16x32_bf16(false, a.v, false, b3, (short)0, acc3, false, false);
    }

    f32x8 accs[4] = { acc0, acc1, acc2, acc3 };
#pragma unroll
    for (int u = 0; u < 4; ++u) {
        const int n = ts * 64 + u * 16 + m16;
        if (n >= N) continue;
        const float bv = bias ? bias[n] : 0.0f;
#pragma unroll
        for (int v = 0; v < 8; ++v) {
            const int m = tm * 16 + v + hi * 8;
            float val = accs[u][v] * scale + bv;
            if (mode == 1)      val = leakyf(val);
            else if (mode == 2) val = fmaxf(val, 0.0f);
            else if (mode == 4) val = addIn[(long)m * addLd + n] + leakyf(val);
            C[(long)m * ldc + n] = val;
        }
    }
}

// ---------------- f32 -> bf16 cast with zero padding --------------------------------
__global__ void cast_pad_bf16(const float* __restrict__ src, int rows, int cols,
                              __bf16* __restrict__ dst, int dRows, int dCols)
{
    long i = (long)blockIdx.x * blockDim.x + threadIdx.x;
    long tot = (long)dRows * dCols;
    if (i >= tot) return;
    int r = (int)(i / dCols), c = (int)(i % dCols);
    float v = (r < rows && c < cols) ? src[(long)r * cols + c] : 0.0f;
    dst[i] = f2bf(v);
}

// ---------------- transpose + cast: dst[r][c] = src[c][r] (zero pad) ----------------
__global__ void transpose_cast_bf16(const float* __restrict__ src, int srcRows, int srcCols,
                                    __bf16* __restrict__ dst, int dRows, int dCols)
{
    long i = (long)blockIdx.x * blockDim.x + threadIdx.x;
    long tot = (long)dRows * dCols;
    if (i >= tot) return;
    int r = (int)(i / dCols), c = (int)(i % dCols);
    float v = (r < srcCols && c < srcRows) ? src[(long)c * srcCols + r] : 0.0f;
    dst[i] = f2bf(v);
}

// ---------------- L2 row-normalize p -> pn (f32) + bf16 padded copy -----------------
__global__ __launch_bounds__(128)
void rownorm(const float* __restrict__ p, float* __restrict__ pn,
             __bf16* __restrict__ pnb, int cols, int dCols)
{
    __shared__ float red[128];
    const int row = blockIdx.x, t = threadIdx.x;
    float v = (t < cols) ? p[(long)row * cols + t] : 0.0f;
    red[t] = v * v;
    __syncthreads();
    for (int s = 64; s > 0; s >>= 1) { if (t < s) red[t] += red[t + s]; __syncthreads(); }
    const float inv = rsqrtf(red[0]);
    const float o = v * inv;
    if (t < cols) pn[(long)row * cols + t] = o;
    pnb[(long)row * dCols + t] = f2bf(t < cols ? o : 0.0f);
}

// ---------------- row sum of squares --------------------------------------------------
__global__ __launch_bounds__(256)
void row_sumsq(const float* __restrict__ x, int cols, float* __restrict__ sq)
{
    __shared__ float red[256];
    const int row = blockIdx.x, t = threadIdx.x;
    float s = 0.0f;
    for (int c = t; c < cols; c += 256) { float v = x[(long)row * cols + c]; s += v * v; }
    red[t] = s; __syncthreads();
    for (int st = 128; st > 0; st >>= 1) { if (t < st) red[t] += red[t + st]; __syncthreads(); }
    if (t == 0) sq[row] = red[0];
}

// ---------------- labels / prefix counts / pairing offsets ----------------------------
__global__ void build_masks(const int* __restrict__ site, int* __restrict__ labels,
                            int* __restrict__ prefixCnt,   // [4][1024]
                            int* __restrict__ rowPosOff, int* __restrict__ rowNegOff,
                            int* __restrict__ meta)        // meta[0]=m
{
    const int j = threadIdx.x;  // 1024 threads
    const int* s = site + j * 4;
    int best = 0, bv = s[0];
    for (int k = 1; k < 4; ++k) if (s[k] > bv) { bv = s[k]; best = k; }
    labels[j] = best;
    __syncthreads();
    if (j == 0) {
        int cnt[4] = {0, 0, 0, 0};
        for (int t = 0; t < BZn; ++t) {
            for (int L = 0; L < 4; ++L) prefixCnt[L * BZn + t] = cnt[L];
            cnt[labels[t]]++;
        }
        int posOff = 0, negOff = 0;
        for (int t = 0; t < BZn; ++t) {
            rowPosOff[t] = posOff; rowNegOff[t] = negOff;
            const int pc = cnt[labels[t]];
            posOff += pc; negOff += BZn - pc;
        }
        meta[1] = posOff; meta[2] = negOff;
        meta[0] = posOff < negOff ? posOff : negOff;
    }
}

// ---------------- scatter sim values into rank-paired pos/neg arrays ------------------
__global__ void scatter_pairs(const float* __restrict__ sim, const int* __restrict__ labels,
                              const int* __restrict__ prefixCnt,
                              const int* __restrict__ rowPosOff, const int* __restrict__ rowNegOff,
                              const int* __restrict__ meta,
                              float* __restrict__ posVal, float* __restrict__ negVal)
{
    long f = (long)blockIdx.x * blockDim.x + threadIdx.x;
    if (f >= (long)BZn * BZn) return;
    const int i = (int)(f >> 10), j = (int)(f & 1023);
    const int Li = labels[i];
    const int sameBefore = prefixCnt[Li * BZn + j];
    const int m = meta[0];
    const float v = sim[f];
    if (labels[j] == Li) {
        const int r = rowPosOff[i] + sameBefore;
        if (r < m) posVal[r] = v;
    } else {
        const int r = rowNegOff[i] + (j - sameBefore);
        if (r < m) negVal[r] = v;
    }
}

// ---------------- loss = mean softplus(neg - pos), two-stage deterministic ------------
__global__ __launch_bounds__(256)
void loss_partial(const float* __restrict__ posVal, const float* __restrict__ negVal,
                  const int* __restrict__ meta, float* __restrict__ part)
{
    __shared__ float red[256];
    const int m = meta[0], t = threadIdx.x;
    float s = 0.0f;
    for (long r = (long)blockIdx.x * 256 + t; r < m; r += (long)gridDim.x * 256) {
        const float x = negVal[r] - posVal[r];
        s += (x > 0.0f) ? x + log1pf(expf(-x)) : log1pf(expf(x));
    }
    red[t] = s; __syncthreads();
    for (int st = 128; st > 0; st >>= 1) { if (t < st) red[t] += red[t + st]; __syncthreads(); }
    if (t == 0) part[blockIdx.x] = red[0];
}

__global__ __launch_bounds__(256)
void loss_final(const float* __restrict__ part, int n, const int* __restrict__ meta,
                float* __restrict__ out)
{
    __shared__ float red[256];
    const int t = threadIdx.x;
    float s = 0.0f;
    for (int i = t; i < n; i += 256) s += part[i];
    red[t] = s; __syncthreads();
    for (int st = 128; st > 0; st >>= 1) { if (t < st) red[t] += red[t + st]; __syncthreads(); }
    if (t == 0) out[0] = red[0] / (float)meta[0];
}

// ---------------- global min/max of sim -----------------------------------------------
__global__ __launch_bounds__(256)
void minmax_partial(const float* __restrict__ sim, long n, float* __restrict__ part)
{
    __shared__ float rmn[256], rmx[256];
    const int t = threadIdx.x;
    float mn = 3.4e38f, mx = -3.4e38f;
    for (long f = (long)blockIdx.x * 256 + t; f < n; f += (long)gridDim.x * 256) {
        const float v = sim[f]; mn = fminf(mn, v); mx = fmaxf(mx, v);
    }
    rmn[t] = mn; rmx[t] = mx; __syncthreads();
    for (int st = 128; st > 0; st >>= 1) {
        if (t < st) { rmn[t] = fminf(rmn[t], rmn[t + st]); rmx[t] = fmaxf(rmx[t], rmx[t + st]); }
        __syncthreads();
    }
    if (t == 0) { part[blockIdx.x * 2] = rmn[0]; part[blockIdx.x * 2 + 1] = rmx[0]; }
}

__global__ __launch_bounds__(256)
void minmax_final(const float* __restrict__ part, int n, float* __restrict__ mnmx)
{
    __shared__ float rmn[256], rmx[256];
    const int t = threadIdx.x;
    float mn = 3.4e38f, mx = -3.4e38f;
    for (int i = t; i < n; i += 256) { mn = fminf(mn, part[i * 2]); mx = fmaxf(mx, part[i * 2 + 1]); }
    rmn[t] = mn; rmx[t] = mx; __syncthreads();
    for (int st = 128; st > 0; st >>= 1) {
        if (t < st) { rmn[t] = fminf(rmn[t], rmn[t + st]); rmx[t] = fmaxf(rmx[t], rmx[t + st]); }
        __syncthreads();
    }
    if (t == 0) { mnmx[0] = rmn[0]; mnmx[1] = rmx[0]; }
}

// ---------------- dist = (exp(-a*sqrt(d2))*(1-sn)+sn) * (cond ? 0.9 : 0.1) ------------
__global__ void build_dist(const float* __restrict__ sim, const float* __restrict__ G,
                           const float* __restrict__ sq, const int* __restrict__ ages,
                           const int* __restrict__ genders, const float* __restrict__ mnmx,
                           float* __restrict__ dist)
{
    long f = (long)blockIdx.x * blockDim.x + threadIdx.x;
    if (f >= (long)BZn * BZn) return;
    const int i = (int)(f >> 10), j = (int)(f & 1023);
    const float mn = mnmx[0], mx = mnmx[1];
    const float sn = (sim[f] - mn) / (mx - mn);
    const float d2 = fmaxf(sq[i] + sq[j] - 2.0f * G[f], 0.0f);
    const float fs = expf(-0.1f * sqrtf(d2));
    const float tot = fs * (1.0f - sn) + sn;
    const int da = ages[i] - ages[j];
    const bool cond = (da <= 2 && da >= -2) && (genders[i] == genders[j]);
    dist[f] = tot * (cond ? 0.9f : 0.1f);
}

// ---------------- per-row top-8 -> sparse adjacency (dense f32 + bf16) -----------------
__global__ __launch_bounds__(256)
void topk_adj(const float* __restrict__ dist, float* __restrict__ adj, __bf16* __restrict__ adjb)
{
    __shared__ float vals[BZn];
    __shared__ float bmax[256];
    __shared__ int   bidx[256];
    const int row = blockIdx.x, t = threadIdx.x;
    for (int c = t; c < BZn; c += 256) {
        vals[c] = dist[(long)row * BZn + c];
        adj[(long)row * BZn + c] = 0.0f;
    }
    __syncthreads();
    for (int kk = 0; kk < TOPKn; ++kk) {
        float mv = -3.4e38f; int mi = BZn;
        for (int c = t; c < BZn; c += 256) {
            const float v = vals[c];
            if (v > mv) { mv = v; mi = c; }
        }
        bmax[t] = mv; bidx[t] = mi; __syncthreads();
        for (int st = 128; st > 0; st >>= 1) {
            if (t < st) {
                if (bmax[t + st] > bmax[t] ||
                    (bmax[t + st] == bmax[t] && bidx[t + st] < bidx[t])) {
                    bmax[t] = bmax[t + st]; bidx[t] = bidx[t + st];
                }
            }
            __syncthreads();
        }
        if (t == 0) { adj[(long)row * BZn + bidx[0]] = bmax[0]; vals[bidx[0]] = -3.4e38f; }
        __syncthreads();
    }
    for (int c = t; c < BZn; c += 256)
        adjb[(long)row * BZn + c] = f2bf(adj[(long)row * BZn + c]);
}

// ---------------- per-column batchnorm stats (biased variance) -------------------------
__global__ __launch_bounds__(256)
void col_stats(const float* __restrict__ x, long rows, int cols,
               float* __restrict__ mean, float* __restrict__ var)
{
    __shared__ float rs[256], rs2[256];
    const int c = blockIdx.x, t = threadIdx.x;
    float s = 0.0f, s2 = 0.0f;
    for (long r = t; r < rows; r += 256) {
        const float v = x[r * cols + c]; s += v; s2 += v * v;
    }
    rs[t] = s; rs2[t] = s2; __syncthreads();
    for (int st = 128; st > 0; st >>= 1) {
        if (t < st) { rs[t] += rs[t + st]; rs2[t] += rs2[t + st]; }
        __syncthreads();
    }
    if (t == 0) {
        const float m = rs[0] / (float)rows;
        mean[c] = m; var[c] = rs2[0] / (float)rows - m * m;
    }
}

__global__ void bn_apply(float* __restrict__ x, const float* __restrict__ mean,
                         const float* __restrict__ var, const float* __restrict__ g,
                         const float* __restrict__ b, long rows, int cols)
{
    long i = (long)blockIdx.x * blockDim.x + threadIdx.x;
    if (i >= rows * (long)cols) return;
    const int c = (int)(i % cols);
    x[i] = (x[i] - mean[c]) * rsqrtf(var[c] + EPSn) * g[c] + b[c];
}

// ---------------- bn3: per-ROI channel over (batch, 8) ---------------------------------
__global__ __launch_bounds__(256)
void bn3_stats(const float* __restrict__ x, float* __restrict__ mean, float* __restrict__ var)
{
    __shared__ float rs[256], rs2[256];
    const int r = blockIdx.x, t = threadIdx.x;
    float s = 0.0f, s2 = 0.0f;
    for (int idx = t; idx < BZn * 8; idx += 256) {
        const int b = idx >> 3, l = idx & 7;
        const float v = x[((long)b * ROIn + r) * 8 + l];
        s += v; s2 += v * v;
    }
    rs[t] = s; rs2[t] = s2; __syncthreads();
    for (int st = 128; st > 0; st >>= 1) {
        if (t < st) { rs[t] += rs[t + st]; rs2[t] += rs2[t + st]; }
        __syncthreads();
    }
    if (t == 0) {
        const float m = rs[0] / (float)(BZn * 8);
        mean[r] = m; var[r] = rs2[0] / (float)(BZn * 8) - m * m;
    }
}

__global__ void bn3_apply(const float* __restrict__ x, float* __restrict__ out,
                          const float* __restrict__ mean, const float* __restrict__ var,
                          const float* __restrict__ g, const float* __restrict__ b)
{
    long i = (long)blockIdx.x * blockDim.x + threadIdx.x;
    if (i >= (long)MRn * 8) return;
    const int r = (int)((i >> 3) % ROIn);
    out[i] = (x[i] - mean[r]) * rsqrtf(var[r] + EPSn) * g[r] + b[r];
}

// =======================================================================================
extern "C" void kernel_launch(void* const* d_in, const int* in_sizes, int n_in,
                              void* d_out, int out_size, void* d_ws, size_t ws_size,
                              hipStream_t stream)
{
    (void)in_sizes; (void)n_in; (void)ws_size;
    const float* x       = (const float*)d_in[0];
    const float* pseudo  = (const float*)d_in[1];
    const int*   ages    = (const int*)d_in[2];
    const int*   genders = (const int*)d_in[3];
    const int*   site    = (const int*)d_in[4];
    const float* fc_p_w  = (const float*)d_in[5];
    const float* fc_p_b  = (const float*)d_in[6];
    const float* proj_w1 = (const float*)d_in[7];
    const float* proj_b1 = (const float*)d_in[8];
    const float* proj_w2 = (const float*)d_in[9];
    const float* proj_b2 = (const float*)d_in[10];
    const float* gcn_w1  = (const float*)d_in[11];
    const float* gcn_b1  = (const float*)d_in[12];
    const float* gcn_w2  = (const float*)d_in[13];
    const float* gcn_b2  = (const float*)d_in[14];
    const float* gcn1_w  = (const float*)d_in[15];
    const float* gcn1_b  = (const float*)d_in[16];
    const float* gcn2_w1 = (const float*)d_in[17];
    const float* gcn2_b1 = (const float*)d_in[18];
    const float* gcn2_w2 = (const float*)d_in[19];
    const float* gcn2_b2 = (const float*)d_in[20];
    const float* bn1_g   = (const float*)d_in[21];
    const float* bn1_b   = (const float*)d_in[22];
    const float* bn2_g   = (const float*)d_in[23];
    const float* bn2_b   = (const float*)d_in[24];
    const float* bn3_g   = (const float*)d_in[25];
    const float* bn3_b   = (const float*)d_in[26];

    float* hOut    = (float*)d_out;
    float* lossOut = (float*)d_out + (out_size - 1);

    // ---- bump allocator over workspace ----
    size_t cur = 0;
    auto alloc = [&](size_t bytes) -> void* {
        cur = (cur + 255) & ~(size_t)255;
        void* p = (char*)d_ws + cur;
        cur += bytes;
        return p;
    };
    float*  xsum   = (float*)alloc((size_t)MRn * DIMn * 4);        // x + leaky(pseudo@W+b); == xf
    float*  FBIG   = (float*)alloc((size_t)MRn * DIMn * 4);        // ping f32 [MR,116] / [1024,13456]
    __bf16* BF1    = (__bf16*)alloc((size_t)MRn * CPn * 2);        // bf16 [MR,128] scratch
    __bf16* Xb     = (__bf16*)alloc((size_t)BZn * FLATP * 2);      // xf bf16 [1024,13472]
    __bf16* BTB    = (__bf16*)alloc((size_t)FLATR * BZn * 2);      // transposed bf16 [13504,1024]
    __bf16* PW1T   = (__bf16*)alloc((size_t)CPn * FLATP * 2);
    __bf16* FcT    = (__bf16*)alloc((size_t)CPn * CPn * 2);
    __bf16* PW2T   = (__bf16*)alloc((size_t)CPn * CPn * 2);
    __bf16* GW1T   = (__bf16*)alloc((size_t)CPn * CPn * 2);
    __bf16* GW2T   = (__bf16*)alloc((size_t)CPn * CPn * 2);
    __bf16* G1T    = (__bf16*)alloc((size_t)CPn * CPn * 2);
    __bf16* W64T   = (__bf16*)alloc((size_t)64 * CPn * 2);
    __bf16* W8T    = (__bf16*)alloc((size_t)64 * 64 * 2);          // 8 real rows, padded to 64
    float*  T1     = (float*)alloc((size_t)BZn * ROIn * 4);
    __bf16* T1b    = (__bf16*)alloc((size_t)BZn * CPn * 2);
    float*  Pf     = (float*)alloc((size_t)BZn * ROIn * 4);
    float*  pn     = (float*)alloc((size_t)BZn * ROIn * 4);
    __bf16* Pnb    = (__bf16*)alloc((size_t)BZn * CPn * 2);
    float*  sim    = (float*)alloc((size_t)BZn * BZn * 4);
    float*  G      = (float*)alloc((size_t)BZn * BZn * 4);
    float*  dist   = (float*)alloc((size_t)BZn * BZn * 4);
    float*  adj    = (float*)alloc((size_t)BZn * BZn * 4);
    __bf16* Adjb   = (__bf16*)alloc((size_t)BZn * BZn * 2);
    float*  posVal = (float*)alloc((size_t)524288 * 4);
    float*  negVal = (float*)alloc((size_t)524288 * 4);
    float*  sq     = (float*)alloc((size_t)BZn * 4);
    int*    labels = (int*)alloc((size_t)BZn * 4);
    int*    pfxCnt = (int*)alloc((size_t)4 * BZn * 4);
    int*    rPosO  = (int*)alloc((size_t)BZn * 4);
    int*    rNegO  = (int*)alloc((size_t)BZn * 4);
    int*    meta   = (int*)alloc(16);
    float*  lpart  = (float*)alloc((size_t)256 * 4);
    float*  mmpart = (float*)alloc((size_t)256 * 2 * 4);
    float*  mnmx   = (float*)alloc(8);
    float*  bnM    = (float*)alloc((size_t)ROIn * 4);
    float*  bnV    = (float*)alloc((size_t)ROIn * 4);

    auto ew = [](long tot) { return (unsigned)((tot + 255) / 256); };
    auto gemm = [&](const __bf16* A, int lda, const __bf16* Bt, int ldb,
                    float* C, int ldc, int M, int N, int Kp,
                    const float* bias, const float* addIn, int addLd,
                    float scale, int mode) {
        long strips = (long)(M / 16) * (((N + 15) / 16 + 3) / 4);
        unsigned blocks = (unsigned)((strips + 7) / 8);
        hipLaunchKernelGGL(wmma_gemm, dim3(blocks), dim3(256), 0, stream,
                           A, lda, Bt, ldb, C, ldc, M, N, Kp, bias, addIn, addLd, scale, mode);
    };

    // --- pairing masks from site labels ---
    hipLaunchKernelGGL(build_masks, dim3(1), dim3(1024), 0, stream,
                       site, labels, pfxCnt, rPosO, rNegO, meta);

    // --- xsum = x + leaky(pseudo @ fc_p_w + b)  (as [MR,116] GEMM) ---
    hipLaunchKernelGGL(cast_pad_bf16, dim3(ew((long)MRn * CPn)), dim3(256), 0, stream,
                       pseudo, MRn, DIMn, BF1, MRn, CPn);
    hipLaunchKernelGGL(transpose_cast_bf16, dim3(ew((long)CPn * CPn)), dim3(256), 0, stream,
                       fc_p_w, ROIn, ROIn, FcT, CPn, CPn);
    gemm(BF1, CPn, FcT, CPn, xsum, ROIn, MRn, ROIn, CPn, fc_p_b, x, DIMn, 1.0f, 4);

    // --- contrastive head ---
    hipLaunchKernelGGL(cast_pad_bf16, dim3(ew((long)BZn * FLATP)), dim3(256), 0, stream,
                       xsum, BZn, FLATn, Xb, BZn, FLATP);
    hipLaunchKernelGGL(transpose_cast_bf16, dim3(ew((long)CPn * FLATP)), dim3(256), 0, stream,
                       proj_w1, FLATn, ROIn, PW1T, CPn, FLATP);
    gemm(Xb, FLATP, PW1T, FLATP, T1, ROIn, BZn, ROIn, FLATP, proj_b1, nullptr, 0, 1.0f, 2);
    hipLaunchKernelGGL(cast_pad_bf16, dim3(ew((long)BZn * CPn)), dim3(256), 0, stream,
                       T1, BZn, ROIn, T1b, BZn, CPn);
    hipLaunchKernelGGL(transpose_cast_bf16, dim3(ew((long)CPn * CPn)), dim3(256), 0, stream,
                       proj_w2, ROIn, ROIn, PW2T, CPn, CPn);
    gemm(T1b, CPn, PW2T, CPn, Pf, ROIn, BZn, ROIn, CPn, proj_b2, nullptr, 0, 1.0f, 0);
    hipLaunchKernelGGL(rownorm, dim3(BZn), dim3(128), 0, stream, Pf, pn, Pnb, ROIn, CPn);

    // sim = (pn @ pn^T) / TEMP   (Gram: A == Bt, no transpose needed)
    gemm(Pnb, CPn, Pnb, CPn, sim, BZn, BZn, BZn, CPn, nullptr, nullptr, 0, INV_TEMP, 0);

    // loss
    hipLaunchKernelGGL(scatter_pairs, dim3(4096), dim3(256), 0, stream,
                       sim, labels, pfxCnt, rPosO, rNegO, meta, posVal, negVal);
    hipLaunchKernelGGL(loss_partial, dim3(256), dim3(256), 0, stream, posVal, negVal, meta, lpart);
    hipLaunchKernelGGL(loss_final, dim3(1), dim3(256), 0, stream, lpart, 256, meta, lossOut);

    // sim min/max
    hipLaunchKernelGGL(minmax_partial, dim3(256), dim3(256), 0, stream,
                       sim, (long)BZn * BZn, mmpart);
    hipLaunchKernelGGL(minmax_final, dim3(1), dim3(256), 0, stream, mmpart, 256, mnmx);

    // Gram of features: G = xf @ xf^T
    hipLaunchKernelGGL(row_sumsq, dim3(BZn), dim3(256), 0, stream, xsum, FLATn, sq);
    gemm(Xb, FLATP, Xb, FLATP, G, BZn, BZn, BZn, FLATP, nullptr, nullptr, 0, 1.0f, 0);

    // dist + top-8 adjacency
    hipLaunchKernelGGL(build_dist, dim3(4096), dim3(256), 0, stream,
                       sim, G, sq, ages, genders, mnmx, dist);
    hipLaunchKernelGGL(topk_adj, dim3(BZn), dim3(256), 0, stream, dist, adj, Adjb);

    // --- GCN stack ---
    // H1 = adj @ xf
    hipLaunchKernelGGL(transpose_cast_bf16, dim3(ew((long)FLATR * BZn)), dim3(256), 0, stream,
                       xsum, BZn, FLATn, BTB, FLATR, BZn);
    gemm(Adjb, BZn, BTB, BZn, FBIG, FLATn, BZn, FLATn, BZn, nullptr, nullptr, 0, 1.0f, 0);
    // h = leaky(H1 @ gcn_w1 + b1) @ gcn_w2 + b2
    hipLaunchKernelGGL(cast_pad_bf16, dim3(ew((long)MRn * CPn)), dim3(256), 0, stream,
                       FBIG, MRn, ROIn, BF1, MRn, CPn);
    hipLaunchKernelGGL(transpose_cast_bf16, dim3(ew((long)CPn * CPn)), dim3(256), 0, stream,
                       gcn_w1, DIMn, ROIn, GW1T, CPn, CPn);
    gemm(BF1, CPn, GW1T, CPn, FBIG, ROIn, MRn, ROIn, CPn, gcn_b1, nullptr, 0, 1.0f, 1);
    hipLaunchKernelGGL(cast_pad_bf16, dim3(ew((long)MRn * CPn)), dim3(256), 0, stream,
                       FBIG, MRn, ROIn, BF1, MRn, CPn);
    hipLaunchKernelGGL(transpose_cast_bf16, dim3(ew((long)CPn * CPn)), dim3(256), 0, stream,
                       gcn_w2, ROIn, ROIn, GW2T, CPn, CPn);
    gemm(BF1, CPn, GW2T, CPn, FBIG, ROIn, MRn, ROIn, CPn, gcn_b2, nullptr, 0, 1.0f, 0);
    // bn1 (in place)
    hipLaunchKernelGGL(col_stats, dim3(ROIn), dim3(256), 0, stream, FBIG, (long)MRn, ROIn, bnM, bnV);
    hipLaunchKernelGGL(bn_apply, dim3(ew((long)MRn * ROIn)), dim3(256), 0, stream,
                       FBIG, bnM, bnV, bn1_g, bn1_b, (long)MRn, ROIn);

    // H2 = adj @ h
    hipLaunchKernelGGL(transpose_cast_bf16, dim3(ew((long)FLATR * BZn)), dim3(256), 0, stream,
                       FBIG, BZn, FLATn, BTB, FLATR, BZn);
    gemm(Adjb, BZn, BTB, BZn, FBIG, FLATn, BZn, FLATn, BZn, nullptr, nullptr, 0, 1.0f, 0);
    // h = leaky(H2 @ gcn1_w + b)
    hipLaunchKernelGGL(cast_pad_bf16, dim3(ew((long)MRn * CPn)), dim3(256), 0, stream,
                       FBIG, MRn, ROIn, BF1, MRn, CPn);
    hipLaunchKernelGGL(transpose_cast_bf16, dim3(ew((long)CPn * CPn)), dim3(256), 0, stream,
                       gcn1_w, ROIn, ROIn, G1T, CPn, CPn);
    gemm(BF1, CPn, G1T, CPn, FBIG, ROIn, MRn, ROIn, CPn, gcn1_b, nullptr, 0, 1.0f, 1);
    // bn2 (in place)
    hipLaunchKernelGGL(col_stats, dim3(ROIn), dim3(256), 0, stream, FBIG, (long)MRn, ROIn, bnM, bnV);
    hipLaunchKernelGGL(bn_apply, dim3(ew((long)MRn * ROIn)), dim3(256), 0, stream,
                       FBIG, bnM, bnV, bn2_g, bn2_b, (long)MRn, ROIn);

    // H3 = adj @ h
    hipLaunchKernelGGL(transpose_cast_bf16, dim3(ew((long)FLATR * BZn)), dim3(256), 0, stream,
                       FBIG, BZn, FLATn, BTB, FLATR, BZn);
    gemm(Adjb, BZn, BTB, BZn, FBIG, FLATn, BZn, FLATn, BZn, nullptr, nullptr, 0, 1.0f, 0);
    // h = leaky(leaky(H3 @ gcn2_w1 + b1) @ gcn2_w2 + b2)
    hipLaunchKernelGGL(cast_pad_bf16, dim3(ew((long)MRn * CPn)), dim3(256), 0, stream,
                       FBIG, MRn, ROIn, BF1, MRn, CPn);
    hipLaunchKernelGGL(transpose_cast_bf16, dim3(ew((long)64 * CPn)), dim3(256), 0, stream,
                       gcn2_w1, ROIn, 64, W64T, 64, CPn);
    gemm(BF1, CPn, W64T, CPn, FBIG, 64, MRn, 64, CPn, gcn2_b1, nullptr, 0, 1.0f, 1);
    hipLaunchKernelGGL(cast_pad_bf16, dim3(ew((long)MRn * 64)), dim3(256), 0, stream,
                       FBIG, MRn, 64, BF1, MRn, 64);
    hipLaunchKernelGGL(transpose_cast_bf16, dim3(ew((long)64 * 64)), dim3(256), 0, stream,
                       gcn2_w2, 64, 8, W8T, 64, 64);
    gemm(BF1, 64, W8T, 64, FBIG, 8, MRn, 8, 64, gcn2_b2, nullptr, 0, 1.0f, 1);

    // bn3 -> output
    hipLaunchKernelGGL(bn3_stats, dim3(ROIn), dim3(256), 0, stream, FBIG, bnM, bnV);
    hipLaunchKernelGGL(bn3_apply, dim3(ew((long)MRn * 8)), dim3(256), 0, stream,
                       FBIG, hOut, bnM, bnV, bn3_g, bn3_b);
}
